// MultiHeadAttention_56186762166614
// MI455X (gfx1250) — matrix operand order repaired and checked by reading, not compiled, not run on hardware
//
#include <hip/hip_runtime.h>
#include <hip/hip_bf16.h>
#include <math.h>

// ---------------- types ----------------
typedef _Float16 h16 __attribute__((ext_vector_type(16)));
typedef _Float16 h8  __attribute__((ext_vector_type(8)));
typedef float    f8  __attribute__((ext_vector_type(8)));
typedef unsigned int u32x4 __attribute__((ext_vector_type(4)));
typedef int i32x8 __attribute__((ext_vector_type(8)));
typedef int i32x4 __attribute__((ext_vector_type(4)));

#define D_MODEL 1024
#define N_HEADS 16
#define HEAD_DIM 64
#define BATCH 4
#define SEQ 2048
#define NTOK (BATCH * SEQ)   // 8192

// ---------------- TDM helper ----------------
// Build a 2D tensor-DMA descriptor (D#) and issue TENSOR_LOAD_TO_LDS.
// tileW is in 4-byte units, tileH in rows, rowStrideUnits = global row stride in
// 4-byte units. padIntervalCode/padAmountCode per D# group1 encodings so the TDM
// writes padded (bank-conflict-free) LDS rows directly.
__device__ __forceinline__ unsigned lds_off(const void* p) {
  return (unsigned)(unsigned long long)(uintptr_t)p;
}

__device__ __forceinline__ void tdm_load_2d(const void* gsrc, unsigned ldsOff,
                                            int tileW, int tileH,
                                            unsigned rowStrideUnits,
                                            int padIntervalCode, int padAmountCode) {
  const unsigned long long ga = (unsigned long long)(uintptr_t)gsrc;
  u32x4 g0;
  g0[0] = 1u;                                                  // count=1 (valid user D#)
  g0[1] = ldsOff;                                              // lds_addr (bytes)
  g0[2] = (unsigned)ga;                                        // global_addr[31:0]
  g0[3] = (unsigned)((ga >> 32) & 0x01FFFFFFu) | (2u << 30);   // addr[56:32] | type=2
  i32x8 g1;
  // data_size=4B (code 2), pad_enable, pad_interval, pad_amount; workgroup_mask=0
  g1[0] = (2 << 16) | (1 << 20) | (padIntervalCode << 22) | (padAmountCode << 25);
  g1[1] = (tileW & 0xFFFF) << 16;                 // tensor_dim0[15:0] (== tileW)
  g1[2] = 0;                                      // tensor_dim0 hi | tensor_dim1 lo (dim1=1<<20)
  g1[3] = 16 | (tileW << 16);                     // tensor_dim1 hi (1<<20 >> 16) | tile_dim0
  g1[4] = tileH & 0xFFFF;                         // tile_dim1 | tile_dim2=0 (2D)
  g1[5] = (int)rowStrideUnits;                    // tensor_dim0_stride[31:0]
  g1[6] = 0;                                      // stride hi | dim1_stride lo (unused, 2D)
  g1[7] = 0;
  const i32x4 z4 = {};
  const i32x8 z8 = {};
  // clang-23 / amdgpu-toolchain 6-arg form: (g0, g1, g2, g3, g4, cpol)
  __builtin_amdgcn_tensor_load_to_lds(g0, g1, z4, z4, z8, 0);
}

// ---------------- fragment loaders ----------------
// A fragment (16x32 f16, M x K): lane m=l&15 holds row m; elements 0..7 -> K=8*hi+j,
// elements 8..15 -> K=16+8*hi+j  (ISA 7.12.2 16-bit A table).
__device__ __forceinline__ h16 load_a_frag(const _Float16* base, int stride, int lane) {
  const int m = lane & 15, hi = lane >> 4;
  const _Float16* r = base + (size_t)m * stride;
  h8 lo = *(const h8*)(r + 8 * hi);
  h8 hh = *(const h8*)(r + 16 + 8 * hi);
  h16 a;
#pragma unroll
  for (int i = 0; i < 8; ++i) { a[i] = lo[i]; a[i + 8] = hh[i]; }
  return a;
}

// B fragment (32x16 f16, K x N) loaded from B^T stored row-major [n][k]:
// lane n=l&15 holds column n; elements 0..7 -> K=8*hi+j, elements 8..15 -> K=16+8*hi+j.
__device__ __forceinline__ h16 load_b_frag(const _Float16* baseT, int stride, int lane) {
  const int n = lane & 15, hi = lane >> 4;
  const _Float16* r = baseT + (size_t)n * stride + 8 * hi;
  h8 lo = *(const h8*)(r);
  h8 hh = *(const h8*)(r + 16);
  h16 b;
#pragma unroll
  for (int i = 0; i < 8; ++i) { b[i] = lo[i]; b[i + 8] = hh[i]; }
  return b;
}

__device__ __forceinline__ f8 wmma_f16(h16 a, h16 b, f8 c) {
  return __builtin_amdgcn_wmma_f32_16x16x32_f16(false, a, false, b, (short)0, c, false, false);
}

// ---------------- fp32 -> f16 convert ----------------
__global__ __launch_bounds__(256) void convert_kernel(const float* __restrict__ src,
                                                      _Float16* __restrict__ dst, int n) {
  int i = (blockIdx.x * 256 + threadIdx.x) * 4;
  if (i + 3 < n) {
    float4 v = *(const float4*)(src + i);
    dst[i + 0] = (_Float16)v.x;
    dst[i + 1] = (_Float16)v.y;
    dst[i + 2] = (_Float16)v.z;
    dst[i + 3] = (_Float16)v.w;
  }
}

// ---------------- projection GEMM (TDM double-buffered, 128x64 block) ----------------
// Y[m][n] = sum_k X[m][k] * W[n][k] + bias[n]
// Each of 4 waves owns 32 rows (2 A-fragments); B-fragments amortized over 8 WMMA.
// mode 0: store f16 to [B,H,S,64]        (Q, K projections)
// mode 1: store f32 row-major [M][1024]  (output projection -> d_out)
// mode 2: store f16 transposed [B,H,64,S] (V projection)
#define LDSA 40  // padded LDS row stride (f16): 64B data + 16B pad, conflict-free
__global__ __launch_bounds__(128) void gemm_kernel(const _Float16* __restrict__ X,
                                                   const _Float16* __restrict__ W,
                                                   const float* __restrict__ bias,
                                                   void* __restrict__ out, int mode) {
  __shared__ _Float16 As[2][128 * LDSA];
  __shared__ _Float16 Ws[2][64 * LDSA];
  const int tid = threadIdx.x;
  const int lane = tid & 31, wave = tid >> 5;
  const int mBase = blockIdx.y * 128;
  const int nBase = blockIdx.x * 64;
  const _Float16* Xb = X + (size_t)mBase * D_MODEL;
  const _Float16* Wb = W + (size_t)nBase * D_MODEL;

  f8 acc[2][4] = {};

  // prime buffer 0 (rows of 64B -> pad interval code 3, pad 16B -> code 3)
  if (wave == 0) {
    tdm_load_2d(Xb, lds_off(&As[0][0]), 16, 128, 512, 3, 3);
    tdm_load_2d(Wb, lds_off(&Ws[0][0]), 16, 64, 512, 3, 3);
    __builtin_amdgcn_s_wait_tensorcnt(0);
  }
  __syncthreads();

  int cur = 0;
  for (int k0 = 0; k0 < D_MODEL; k0 += 32) {
    if ((k0 + 32 < D_MODEL) && wave == 0) {
      tdm_load_2d(Xb + k0 + 32, lds_off(&As[cur ^ 1][0]), 16, 128, 512, 3, 3);
      tdm_load_2d(Wb + k0 + 32, lds_off(&Ws[cur ^ 1][0]), 16, 64, 512, 3, 3);
    }

    // issue all fragment loads first, then the WMMA chain (partial DScnt waits)
    const _Float16* Ac = &As[cur][0] + wave * 32 * LDSA;
    const _Float16* Wc = &Ws[cur][0];
    const h16 a0 = load_a_frag(Ac, LDSA, lane);
    const h16 a1 = load_a_frag(Ac + 16 * LDSA, LDSA, lane);
    const h16 b0 = load_b_frag(Wc + 0 * 16 * LDSA, LDSA, lane);
    const h16 b1 = load_b_frag(Wc + 1 * 16 * LDSA, LDSA, lane);
    const h16 b2 = load_b_frag(Wc + 2 * 16 * LDSA, LDSA, lane);
    const h16 b3 = load_b_frag(Wc + 3 * 16 * LDSA, LDSA, lane);
    acc[0][0] = wmma_f16(a0, b0, acc[0][0]);
    acc[0][1] = wmma_f16(a0, b1, acc[0][1]);
    acc[0][2] = wmma_f16(a0, b2, acc[0][2]);
    acc[0][3] = wmma_f16(a0, b3, acc[0][3]);
    acc[1][0] = wmma_f16(a1, b0, acc[1][0]);
    acc[1][1] = wmma_f16(a1, b1, acc[1][1]);
    acc[1][2] = wmma_f16(a1, b2, acc[1][2]);
    acc[1][3] = wmma_f16(a1, b3, acc[1][3]);

    if (wave == 0) __builtin_amdgcn_s_wait_tensorcnt(0);
    __syncthreads();
    cur ^= 1;
  }

  const int n15 = lane & 15, hi = lane >> 4;
#pragma unroll
  for (int mi = 0; mi < 2; ++mi) {
    const int mRow0 = mBase + wave * 32 + mi * 16 + hi * 8;  // rows mRow0 .. mRow0+7
#pragma unroll
    for (int nt = 0; nt < 4; ++nt) {
      const int col = nBase + nt * 16 + n15;
      const float bv = bias[col];
      if (mode == 1) {
        float* o = (float*)out;
#pragma unroll
        for (int r = 0; r < 8; ++r)
          o[(size_t)(mRow0 + r) * D_MODEL + col] = acc[mi][nt][r] + bv;
      } else {
        _Float16* o = (_Float16*)out;
        const int hh = col >> 6, dh = col & 63;
        const int bb = mRow0 >> 11, s0 = mRow0 & 2047;
        if (mode == 0) {
          const size_t base = ((size_t)bb * N_HEADS + hh) * SEQ;
#pragma unroll
          for (int r = 0; r < 8; ++r)
            o[(base + s0 + r) * HEAD_DIM + dh] = (_Float16)(acc[mi][nt][r] + bv);
        } else {  // mode 2: Vt [B,H,64,S], 8 consecutive s -> one b128 store
          h8 vv;
#pragma unroll
          for (int r = 0; r < 8; ++r) vv[r] = (_Float16)(acc[mi][nt][r] + bv);
          *(h8*)(o + (((size_t)bb * N_HEADS + hh) * HEAD_DIM + dh) * SEQ + s0) = vv;
        }
      }
    }
  }
}

// ---------------- flash attention (TDM double-buffered K/V) ----------------
// grid: (S/64, H, B); block: 128 threads (4 waves). Each wave owns 16 q rows.
#define LDSK 72  // padded LDS row stride: 128B data + 16B pad, conflict-free
__global__ __launch_bounds__(128) void attn_kernel(const _Float16* __restrict__ Q,
                                                   const _Float16* __restrict__ K,
                                                   const _Float16* __restrict__ Vt,
                                                   const int* __restrict__ mask,
                                                   _Float16* __restrict__ ctx) {
  __shared__ _Float16 Ks[2][64 * LDSK];     // rows j (64), cols d (64)
  __shared__ _Float16 Vs[2][64 * LDSK];     // rows dh (64), cols j (64)
  __shared__ _Float16 Ps[4 * 16 * LDSK];    // per-wave P tile: rows m (16), cols j (64)

  const int tid = threadIdx.x, lane = tid & 31, wave = tid >> 5;
  const int m15 = lane & 15, hi = lane >> 4;
  const int qBase = blockIdx.x * 64;
  const int h = blockIdx.y, b = blockIdx.z;
  const size_t bh = (size_t)b * N_HEADS + h;

  const _Float16* Qp = Q + bh * SEQ * HEAD_DIM;
  const _Float16* Kp = K + bh * SEQ * HEAD_DIM;
  const _Float16* Vp = Vt + bh * HEAD_DIM * SEQ;

  // Q fragments for this wave's 16 rows, kept in registers the whole kernel.
  const _Float16* qtile = Qp + (size_t)(qBase + wave * 16) * HEAD_DIM;
  const h16 qf0 = load_a_frag(qtile, HEAD_DIM, lane);       // d 0..31
  const h16 qf1 = load_a_frag(qtile + 32, HEAD_DIM, lane);  // d 32..63

  f8 o0 = {}, o1 = {}, o2 = {}, o3 = {};
  float mrow[8], lrow[8];
#pragma unroll
  for (int r = 0; r < 8; ++r) { mrow[r] = -1e30f; lrow[r] = 0.0f; }
  const float scale = 0.03125f;  // 1/sqrt(1024)

  _Float16* Pw = Ps + wave * 16 * LDSK;

  // prime buffer 0: K chunk rows 128B (stride 32 units), V chunk rows stride S f16
  if (wave == 0) {
    tdm_load_2d(Kp, lds_off(&Ks[0][0]), 32, 64, 32, 4, 3);
    tdm_load_2d(Vp, lds_off(&Vs[0][0]), 32, 64, 1024, 4, 3);
    __builtin_amdgcn_s_wait_tensorcnt(0);
  }
  __syncthreads();

  int cur = 0;
  for (int j0 = 0; j0 < SEQ; j0 += 64) {
    if ((j0 + 64 < SEQ) && wave == 0) {
      tdm_load_2d(Kp + (size_t)(j0 + 64) * HEAD_DIM, lds_off(&Ks[cur ^ 1][0]), 32, 64, 32, 4, 3);
      tdm_load_2d(Vp + (j0 + 64), lds_off(&Vs[cur ^ 1][0]), 32, 64, 1024, 4, 3);
    }
    const _Float16* Kc = &Ks[cur][0];
    const _Float16* Vc = &Vs[cur][0];

    // ---- all four 16x16 score tiles for this 64-wide chunk ----
    f8 st[4];
#pragma unroll
    for (int jt = 0; jt < 4; ++jt) {
      f8 s = {};
      s = wmma_f16(qf0, load_b_frag(Kc + jt * 16 * LDSK, LDSK, lane), s);
      s = wmma_f16(qf1, load_b_frag(Kc + jt * 16 * LDSK + 32, LDSK, lane), s);
      const int jcol = j0 + jt * 16 + m15;
      const bool mz = (mask[(size_t)b * SEQ + jcol] == 0);
#pragma unroll
      for (int r = 0; r < 8; ++r) st[jt][r] = mz ? -1e30f : s[r] * scale;
    }

    // ---- one online-softmax update per chunk (fold 4 tiles elementwise first) ----
#pragma unroll
    for (int r = 0; r < 8; ++r) {
      float v = fmaxf(fmaxf(st[0][r], st[1][r]), fmaxf(st[2][r], st[3][r]));
      v = fmaxf(v, __shfl_xor(v, 1));
      v = fmaxf(v, __shfl_xor(v, 2));
      v = fmaxf(v, __shfl_xor(v, 4));
      v = fmaxf(v, __shfl_xor(v, 8));
      const float mn = fmaxf(mrow[r], v);
      const float corr = __expf(mrow[r] - mn);
      const float p0 = __expf(st[0][r] - mn);
      const float p1 = __expf(st[1][r] - mn);
      const float p2 = __expf(st[2][r] - mn);
      const float p3 = __expf(st[3][r] - mn);
      float sum = (p0 + p1) + (p2 + p3);
      sum += __shfl_xor(sum, 1);
      sum += __shfl_xor(sum, 2);
      sum += __shfl_xor(sum, 4);
      sum += __shfl_xor(sum, 8);
      lrow[r] = lrow[r] * corr + sum;
      mrow[r] = mn;
      o0[r] *= corr; o1[r] *= corr; o2[r] *= corr; o3[r] *= corr;
      // stash p in C-layout position; re-read below in A-layout (same-wave LDS is in-order)
      _Float16* pd = Pw + (r + 8 * hi) * LDSK + m15;
      pd[0]  = (_Float16)p0;
      pd[16] = (_Float16)p1;
      pd[32] = (_Float16)p2;
      pd[48] = (_Float16)p3;
    }

    // ---- ctx += P(16x64) x V(64x64) ----
#pragma unroll
    for (int kc = 0; kc < 2; ++kc) {
      const h16 pa = load_a_frag(Pw + kc * 32, LDSK, lane);
      o0 = wmma_f16(pa, load_b_frag(Vc + 0 * 16 * LDSK + kc * 32, LDSK, lane), o0);
      o1 = wmma_f16(pa, load_b_frag(Vc + 1 * 16 * LDSK + kc * 32, LDSK, lane), o1);
      o2 = wmma_f16(pa, load_b_frag(Vc + 2 * 16 * LDSK + kc * 32, LDSK, lane), o2);
      o3 = wmma_f16(pa, load_b_frag(Vc + 3 * 16 * LDSK + kc * 32, LDSK, lane), o3);
    }

    if (wave == 0) __builtin_amdgcn_s_wait_tensorcnt(0);
    __syncthreads();
    cur ^= 1;
  }

  // normalize and write ctx in concat layout [B,S,D_MODEL] (f16), ready as GEMM A.
  float inv[8];
#pragma unroll
  for (int r = 0; r < 8; ++r) inv[r] = 1.0f / lrow[r];
  const int s0 = qBase + wave * 16 + 8 * hi;
  const size_t obase = (size_t)b * SEQ * D_MODEL + (size_t)h * HEAD_DIM;
  f8 os[4] = {o0, o1, o2, o3};
#pragma unroll
  for (int nt = 0; nt < 4; ++nt) {
    const int dh = nt * 16 + m15;
#pragma unroll
    for (int r = 0; r < 8; ++r)
      ctx[obase + (size_t)(s0 + r) * D_MODEL + dh] = (_Float16)(os[nt][r] * inv[r]);
  }
}

// ---------------- launch ----------------
extern "C" void kernel_launch(void* const* d_in, const int* in_sizes, int n_in,
                              void* d_out, int out_size, void* d_ws, size_t ws_size,
                              hipStream_t stream) {
  const float* q   = (const float*)d_in[0];
  const float* k   = (const float*)d_in[1];
  const float* v   = (const float*)d_in[2];
  const int*   msk = (const int*)d_in[3];
  const float* Wq  = (const float*)d_in[4];
  const float* bq  = (const float*)d_in[5];
  const float* Wk  = (const float*)d_in[6];
  const float* bk  = (const float*)d_in[7];
  const float* Wv  = (const float*)d_in[8];
  const float* bv  = (const float*)d_in[9];
  const float* Wo  = (const float*)d_in[10];
  const float* bo  = (const float*)d_in[11];
  float* out = (float*)d_out;

  char* ws = (char*)d_ws;
  size_t off = 0;
  auto alloc = [&](size_t bytes) -> _Float16* {
    _Float16* p = (_Float16*)(ws + off);
    off += (bytes + 255) & ~(size_t)255;
    return p;
  };
  const size_t TOKB = (size_t)NTOK * D_MODEL * sizeof(_Float16);  // 16 MB
  const size_t WB   = (size_t)D_MODEL * D_MODEL * sizeof(_Float16);

  _Float16* qh  = alloc(TOKB);
  _Float16* kh  = alloc(TOKB);
  _Float16* vh  = alloc(TOKB);
  _Float16* wqh = alloc(WB);
  _Float16* wkh = alloc(WB);
  _Float16* wvh = alloc(WB);
  _Float16* woh = alloc(WB);
  _Float16* Qp  = alloc(TOKB);  // [B,H,S,64]
  _Float16* Kp  = alloc(TOKB);  // [B,H,S,64]
  _Float16* Vt  = alloc(TOKB);  // [B,H,64,S]
  _Float16* ctx = alloc(TOKB);  // [B,S,D]

  const int nTok = NTOK * D_MODEL;   // 8388608
  const int nW = D_MODEL * D_MODEL;  // 1048576
  convert_kernel<<<nTok / 1024, 256, 0, stream>>>(q, qh, nTok);
  convert_kernel<<<nTok / 1024, 256, 0, stream>>>(k, kh, nTok);
  convert_kernel<<<nTok / 1024, 256, 0, stream>>>(v, vh, nTok);
  convert_kernel<<<nW / 1024, 256, 0, stream>>>(Wq, wqh, nW);
  convert_kernel<<<nW / 1024, 256, 0, stream>>>(Wk, wkh, nW);
  convert_kernel<<<nW / 1024, 256, 0, stream>>>(Wv, wvh, nW);
  convert_kernel<<<nW / 1024, 256, 0, stream>>>(Wo, woh, nW);

  dim3 ggrid(D_MODEL / 64, NTOK / 128);  // (16, 64)
  gemm_kernel<<<ggrid, 128, 0, stream>>>(qh, wqh, bq, (void*)Qp, 0);
  gemm_kernel<<<ggrid, 128, 0, stream>>>(kh, wkh, bk, (void*)Kp, 0);
  gemm_kernel<<<ggrid, 128, 0, stream>>>(vh, wvh, bv, (void*)Vt, 2);

  dim3 agrid(SEQ / 64, N_HEADS, BATCH);  // (32, 16, 4)
  attn_kernel<<<agrid, 128, 0, stream>>>(Qp, Kp, Vt, msk, ctx);

  gemm_kernel<<<ggrid, 128, 0, stream>>>(ctx, woh, bo, (void*)out, 1);
}
